// YatConvTranspose1D_19791209300377
// MI455X (gfx1250) — matrix operand order repaired and checked by reading, not compiled, not run on hardware
//
#include <hip/hip_runtime.h>
#include <math.h>

// ---------------------------------------------------------------------------
// YatConvTranspose1D for MI455X (gfx1250), bf16x3 split-precision WMMA GEMM.
//   B=32, L=2048, Cin=256, W=4, F=256, stride=2, dilation=1 -> Lo=4098
// Output q = 2m + r gets taps  x[m]·K[r]  +  x[m-1]·K[r+2]   (zero-padded).
// ---------------------------------------------------------------------------

typedef __attribute__((ext_vector_type(16))) __bf16 v16bf;
typedef __attribute__((ext_vector_type(8)))  __bf16 v8bf;
typedef __attribute__((ext_vector_type(8)))  float  v8f;
typedef __attribute__((ext_vector_type(4)))  unsigned short us4;

#define NBATCH 32
#define LIN    2048
#define CIN    256
#define NF     256
#define LOUT   4098

#define MB     32            // m rows per workgroup
#define NB     64            // filters per workgroup
#define XROWS  (MB + 1)      // rows m0-1 .. m0+31
#define XP     (CIN + 8)     // LDS row pitch (bf16 elems), padded vs bank conflicts
#define BP     (2 * CIN + 8) // LDS B row pitch over K=512, padded

// LDS layout in ushort (bf16-bit) units
#define OFF_XH 0
#define OFF_XL (OFF_XH + XROWS * XP)
#define OFF_BH (OFF_XL + XROWS * XP)
#define OFF_BL (OFF_BH + 2 * NB * BP)
#define OFF_END (OFF_BL + 2 * NB * BP)                 // 150544 ushorts
#define SMEM_BYTES (OFF_END * 2 + (XROWS + NB) * 4 + 32)  // ~301.5 KB

// round-to-nearest-even f32 -> bf16 bits, plus residual lo part
struct HL { unsigned short h, l; };
__device__ __forceinline__ HL bf_split(float f) {
    unsigned u  = __float_as_uint(f);
    unsigned r  = u + 0x7fffu + ((u >> 16) & 1u);
    unsigned short h = (unsigned short)(r >> 16);
    float hf = __uint_as_float((r >> 16) << 16);
    float lo = f - hf;
    unsigned u2 = __float_as_uint(lo);
    unsigned r2 = u2 + 0x7fffu + ((u2 >> 16) & 1u);
    HL out;
    out.h = h;
    out.l = (unsigned short)(r2 >> 16);
    return out;
}

// A fragment (16x32 bf16): per lane one contiguous 16-elem K run
__device__ __forceinline__ v16bf ldA(const unsigned short* p) {
    v8bf a = *reinterpret_cast<const v8bf*>(p);
    v8bf b = *reinterpret_cast<const v8bf*>(p + 8);
    return __builtin_shufflevector(a, b, 0,1,2,3,4,5,6,7,8,9,10,11,12,13,14,15);
}
// B fragment (32x16 bf16): per lane one contiguous 16-elem K run
__device__ __forceinline__ v16bf ldB(const unsigned short* p) {
    v8bf a = *reinterpret_cast<const v8bf*>(p);
    v8bf b = *reinterpret_cast<const v8bf*>(p + 8);
    return __builtin_shufflevector(a, b, 0,1,2,3,4,5,6,7,8,9,10,11,12,13,14,15);
}

__launch_bounds__(256)
__global__ void yat_convT1d_wmma(const float* __restrict__ x,
                                 const float* __restrict__ kern,
                                 const float* __restrict__ bias,
                                 const float* __restrict__ alpha,
                                 float* __restrict__ out) {
    extern __shared__ unsigned char smem_raw[];
    unsigned short* xh = (unsigned short*)smem_raw;
    unsigned short* xl = xh + (OFF_XL - OFF_XH);
    unsigned short* bh = xh + OFF_BH;
    unsigned short* bl = xh + OFF_BL;
    float* s_lds   = (float*)(smem_raw + OFF_END * 2);  // [XROWS] sum(x^2) per row
    float* ksq_lds = s_lds + XROWS;                     // [NB] sum(k^2) per filter

    const int tid  = threadIdx.x;
    const int lane = tid & 31;
    const int wave = tid >> 5;
    const int bidx = blockIdx.z;
    const int m0   = blockIdx.y * MB;   // m in [0, 2049)
    const int n0   = blockIdx.x * NB;

    if (tid < XROWS) s_lds[tid] = 0.0f;
    if (tid < NB)    ksq_lds[tid] = 0.0f;
    __syncthreads();

    // ---- Stage x rows m0-1 .. m0+31 (f32 -> bf16 hi/lo), accumulate s[m] ----
    for (int c = tid; c < XROWS * (CIN / 4); c += 256) {
        int r  = c >> 6;            // local row 0..32
        int c4 = c & 63;            // float4 column
        int gm = m0 - 1 + r;
        float4 v = make_float4(0.f, 0.f, 0.f, 0.f);
        if (gm >= 0 && gm < LIN)
            v = *((const float4*)(x + ((size_t)bidx * LIN + gm) * CIN) + c4);
        HL s0 = bf_split(v.x), s1 = bf_split(v.y), s2 = bf_split(v.z), s3 = bf_split(v.w);
        us4 hi = {s0.h, s1.h, s2.h, s3.h};
        us4 lo = {s0.l, s1.l, s2.l, s3.l};
        *(us4*)(xh + r * XP + 4 * c4) = hi;
        *(us4*)(xl + r * XP + 4 * c4) = lo;
        atomicAdd(&s_lds[r], v.x*v.x + v.y*v.y + v.z*v.z + v.w*v.w);
    }

    // ---- Stage kernel slices for our 64 filters, accumulate ksq[f] ----
    // B[par][n][k<256]=kernel[par][n][k];  B[par][n][k>=256]=kernel[par+2][n][k-256]
    for (int c = tid; c < 4 * NB * (CIN / 4); c += 256) {   // 16384 chunks
        int w   = c >> 12;
        int rem = c & 4095;
        int nl  = rem >> 6;
        int c4  = rem & 63;
        float4 v = *((const float4*)(kern + ((size_t)w * NF + (n0 + nl)) * CIN) + c4);
        HL s0 = bf_split(v.x), s1 = bf_split(v.y), s2 = bf_split(v.z), s3 = bf_split(v.w);
        us4 hi = {s0.h, s1.h, s2.h, s3.h};
        us4 lo = {s0.l, s1.l, s2.l, s3.l};
        int par   = w & 1;
        int koff  = (w >> 1) * CIN + 4 * c4;     // position in K=512
        *(us4*)(bh + (par * NB + nl) * BP + koff) = hi;
        *(us4*)(bl + (par * NB + nl) * BP + koff) = lo;
        atomicAdd(&ksq_lds[nl], v.x*v.x + v.y*v.y + v.z*v.z + v.w*v.w);
    }
    __syncthreads();

    // ---- Main GEMM: wave owns (m-tile, n-tile) x both parities ----
    const int mt    = wave >> 2;         // 0..1
    const int nt    = wave & 3;          // 0..3
    const int mrow  = lane & 15;         // A: M index / B,C,D: N index
    const int khalf = lane >> 4;

    v8f acc_e = {0.f,0.f,0.f,0.f,0.f,0.f,0.f,0.f};
    v8f acc_o = {0.f,0.f,0.f,0.f,0.f,0.f,0.f,0.f};

    const int bn = (nt * 16 + mrow) * BP;            // B row base for this lane
    #pragma unroll
    for (int ks = 0; ks < 16; ++ks) {
        const int kb   = ks * 32;
        // A: K<256 -> x[m] (local row ml+1); K>=256 -> x[m-1] (local row ml)
        const int arow = mt * 16 + mrow + ((kb < 256) ? 1 : 0);
        const int ak   = (kb & 255) + khalf * 16;
        v16bf Ah = ldA(xh + arow * XP + ak);
        v16bf Al = ldA(xl + arow * XP + ak);
        const int bk = kb + khalf * 16;
        v16bf Bhe = ldB(bh + bn + bk);
        v16bf Bho = ldB(bh + NB * BP + bn + bk);
        v16bf Ble = ldB(bl + bn + bk);
        v16bf Blo = ldB(bl + NB * BP + bn + bk);
        // bf16x3: hi*hi + lo*hi + hi*lo, f32 accumulate
        acc_e = __builtin_amdgcn_wmma_f32_16x16x32_bf16(false, Ah, false, Bhe, (short)0, acc_e, false, false);
        acc_o = __builtin_amdgcn_wmma_f32_16x16x32_bf16(false, Ah, false, Bho, (short)0, acc_o, false, false);
        acc_e = __builtin_amdgcn_wmma_f32_16x16x32_bf16(false, Al, false, Bhe, (short)0, acc_e, false, false);
        acc_o = __builtin_amdgcn_wmma_f32_16x16x32_bf16(false, Al, false, Bho, (short)0, acc_o, false, false);
        acc_e = __builtin_amdgcn_wmma_f32_16x16x32_bf16(false, Ah, false, Ble, (short)0, acc_e, false, false);
        acc_o = __builtin_amdgcn_wmma_f32_16x16x32_bf16(false, Ah, false, Blo, (short)0, acc_o, false, false);
    }

    // ---- Epilogue: yat transform + bias + scale, then store ----
    const int   fg    = n0 + nt * 16 + mrow;         // this lane's filter (N col)
    const float ksqv  = ksq_lds[nt * 16 + mrow];
    const float bv    = bias[fg];
    const float a0    = alpha[0];
    const float scale = powf(16.0f / log1pf(256.0f), a0);   // (sqrt(F)/log1p(F))^alpha

    #pragma unroll
    for (int j = 0; j < 8; ++j) {
        const int ml = mt * 16 + khalf * 8 + j;      // C/D: lanes>=16 hold M=8+j
        const int m  = m0 + ml;
        if (m <= LIN) {                              // m in [0, 2048]
            const float psq  = s_lds[ml] + s_lds[ml + 1];   // s[m-1] + s[m]
            const size_t base = ((size_t)bidx * LOUT + 2 * m) * NF + fg;
            const float d0 = acc_e[j];
            const float t0 = psq + ksqv - 2.0f * d0;
            out[base]      = (d0 * d0 / (t0 + 1e-5f) + bv) * scale;
            const float d1 = acc_o[j];
            const float t1 = psq + ksqv - 2.0f * d1;
            out[base + NF] = (d1 * d1 / (t1 + 1e-5f) + bv) * scale;
        }
    }
}

extern "C" void kernel_launch(void* const* d_in, const int* in_sizes, int n_in,
                              void* d_out, int out_size, void* d_ws, size_t ws_size,
                              hipStream_t stream) {
    (void)in_sizes; (void)n_in; (void)out_size; (void)d_ws; (void)ws_size;
    const float* x     = (const float*)d_in[0];
    const float* kern  = (const float*)d_in[1];
    const float* bias  = (const float*)d_in[2];
    const float* alpha = (const float*)d_in[3];
    float* out = (float*)d_out;

    dim3 grid(NF / NB, (2049 + MB - 1) / MB, NBATCH);   // (4, 65, 32)
    dim3 block(256);
    hipLaunchKernelGGL(yat_convT1d_wmma, grid, block, SMEM_BYTES, stream,
                       x, kern, bias, alpha, out);
}
